// Sample_SparseMoE_57509612093490
// MI455X (gfx1250) — matrix operand order repaired and checked by reading, not compiled, not run on hardware
//
#include <hip/hip_runtime.h>
#include <hip/hip_bf16.h>
#include <math.h>

// Problem constants (from reference)
#define B_    32
#define S_    512
#define D_    1024
#define H_    4096
#define E_    8
#define TOPK_ 2
#define CAP_  8          // max(int(B*TOPK/E * CF), 1) = 8

// Fused-kernel tiling
#define TILE_S 32
#define HCHUNK 2048
#define PX (D_ + 8)          // x_lds pitch (bf16 elements) -> bank-conflict-free frags
#define PH (HCHUNK + 8)      // h_lds pitch
#define SMEM_BYTES ((TILE_S * PX + TILE_S * PH) * 2)   // 197632 B < 320 KB WGP LDS

// Workspace layout (bytes). Requires ws_size >= ~161 MB.
#define WS_RI     0u
#define WS_PAIR_E 131072u
#define WS_PAIR_W 131328u
#define WS_XBF    262144u
#define WS_W1T    (262144u + 33554432u)                  // x bf16 = 32 MB
#define WS_W2T    (262144u + 33554432u + 67108864u)      // each Wt bf16 = 64 MB

typedef __attribute__((ext_vector_type(16))) __bf16 bf16x16;
typedef __attribute__((ext_vector_type(8)))  __bf16 bf16x8;
typedef __attribute__((ext_vector_type(8)))  float  v8f;

// ---------------- WMMA fragment helpers (wave32, bf16 16x16x32) ----------------
// A-matrix (16x32, 16-bit): lane l holds row M=l&15; K = kbase + (l>>4)*8 .. +7
// in VGPR0-3 and +16..+23 in VGPR4-7  -> two contiguous 16B chunks, 32B apart.
__device__ __forceinline__ bf16x16 frag_a(const __bf16* p) {
  bf16x8 lo = *(const bf16x8*)p;
  bf16x8 hi = *(const bf16x8*)(p + 16);
  return __builtin_shufflevector(lo, hi, 0,1,2,3,4,5,6,7,8,9,10,11,12,13,14,15);
}
// B-matrix (32x16, 16-bit): lane l holds col N=l&15; K = kbase + (l>>4)*8 .. +7
// and +16..+23 -> two contiguous 16B chunks, 32B apart (same as A).
__device__ __forceinline__ bf16x16 frag_b(const __bf16* p) {
  bf16x8 lo = *(const bf16x8*)p;
  bf16x8 hi = *(const bf16x8*)(p + 16);
  return __builtin_shufflevector(lo, hi, 0,1,2,3,4,5,6,7,8,9,10,11,12,13,14,15);
}
__device__ __forceinline__ v8f wmma_bf16(bf16x16 a, bf16x16 b, v8f c) {
  return __builtin_amdgcn_wmma_f32_16x16x32_bf16(false, a, false, b, (short)0, c,
                                                 false, false);
}
__device__ __forceinline__ float gelu_exact(float z) {
  return 0.5f * z * (1.0f + erff(z * 0.7071067811865476f));
}

// ---------------- 1) ri = mean_s(router_input) ----------------
__global__ void moe_mean_kernel(const float* __restrict__ rin, float* __restrict__ ri) {
  const int b = blockIdx.x;
  const float* base = rin + (size_t)b * S_ * D_;
  for (int d = threadIdx.x; d < D_; d += 256) {
    float acc = 0.f;
    for (int s = 0; s < S_; ++s) acc += base[(size_t)s * D_ + d];
    ri[b * D_ + d] = acc * (1.0f / (float)S_);
  }
}

// ---------------- 2) router: noisy top-2, gate softmax, capacity keep ----------------
__global__ void moe_router_kernel(const float* __restrict__ ri, const float* __restrict__ noise,
                                  const float* __restrict__ Wg, const float* __restrict__ bg,
                                  const float* __restrict__ Wn, const float* __restrict__ bn,
                                  int* __restrict__ pair_e, float* __restrict__ pair_w) {
  __shared__ int   se[B_ * TOPK_];
  __shared__ float sw[B_ * TOPK_];
  const int b = threadIdx.x;            // 32 threads = 1 wave, one sample each
  float lg[E_], nl[E_];
  #pragma unroll
  for (int e = 0; e < E_; ++e) { lg[e] = bg[e]; nl[e] = bn[e]; }
  const float* r = ri + b * D_;
  for (int d = 0; d < D_; ++d) {
    const float rv = r[d];
    #pragma unroll
    for (int e = 0; e < E_; ++e) {
      lg[e] += rv * Wg[d * E_ + e];
      nl[e] += rv * Wn[d * E_ + e];
    }
  }
  float ns[E_];
  #pragma unroll
  for (int e = 0; e < E_; ++e) {
    const float x  = nl[e];
    const float sp = (x > 20.f) ? x : log1pf(expf(x));    // softplus
    ns[e] = lg[e] + noise[b * E_ + e] * sp;
  }
  int i0 = 0; float v0 = ns[0];
  #pragma unroll
  for (int e = 1; e < E_; ++e) if (ns[e] > v0) { v0 = ns[e]; i0 = e; }   // first max (jax tie rule)
  int i1 = -1; float v1 = -3.402823e38f;
  #pragma unroll
  for (int e = 0; e < E_; ++e) if (e != i0 && ns[e] > v1) { v1 = ns[e]; i1 = e; }
  const float t  = expf(v1 - v0);        // v0 >= v1 -> stable
  const float g0 = 1.0f / (1.0f + t);
  const float g1 = t * g0;
  se[b * 2 + 0] = i0; se[b * 2 + 1] = i1;
  sw[b * 2 + 0] = g0; sw[b * 2 + 1] = g1;
  __syncthreads();
  if (b == 0) {                          // stable-order capacity drop (flat pair order)
    int cnt[E_];
    #pragma unroll
    for (int e = 0; e < E_; ++e) cnt[e] = 0;
    for (int p = 0; p < B_ * TOPK_; ++p) {
      const int e = se[p];
      const float w = (cnt[e] < CAP_) ? sw[p] : 0.f;
      cnt[e]++;
      pair_e[p] = e;
      pair_w[p] = w;
    }
  }
}

// ---------------- 3) fp32 -> bf16 elementwise ----------------
__global__ void moe_cvt_bf16_kernel(const float* __restrict__ src, __bf16* __restrict__ dst,
                                    size_t n) {
  size_t i = (size_t)blockIdx.x * blockDim.x + threadIdx.x;
  const size_t stride = (size_t)gridDim.x * blockDim.x;
  for (; i < n; i += stride) dst[i] = (__bf16)src[i];
}

// ---------------- 4) per-expert transpose + convert: dst[c][r] = src[r][c] ----------------
__global__ void moe_transpose_cvt_kernel(const float* __restrict__ src, __bf16* __restrict__ dst,
                                         int R, int C) {
  __shared__ float tile[32][33];
  const int e  = blockIdx.z;
  const int c0 = blockIdx.x * 32;
  const int r0 = blockIdx.y * 32;
  const float* s = src + (size_t)e * R * C;
  __bf16*      d = dst + (size_t)e * R * C;
  const int tx = threadIdx.x, ty = threadIdx.y;   // 32 x 8
  #pragma unroll
  for (int i = 0; i < 4; ++i)
    tile[ty + i * 8][tx] = s[(size_t)(r0 + ty + i * 8) * C + (c0 + tx)];
  __syncthreads();
  #pragma unroll
  for (int i = 0; i < 4; ++i)
    d[(size_t)(c0 + ty + i * 8) * R + (r0 + tx)] = (__bf16)tile[tx][ty + i * 8];
}

// ---------------- 5) fused MoE expert MLP (bf16 WMMA, f32 accumulate) ----------------
// Grid: (S/TILE_S, B). Each WG handles one sample's 32-row s-tile for BOTH top-2 pairs.
// GEMM1 computes h^T tiles (M = h-col, N = s): each wave owns a full h-tile and reuses
// one W1^T A-fragment for BOTH s-tiles; lane's 8 accumulator values are consecutive
// h-columns -> single ds_store_b128 per s-tile after bias + exact GELU + gate scale.
// GEMM2: kb-outer loop loads the two h A-fragments once and reuses each W2^T
// B-fragment for both row-blocks; 128 persistent f32 acc VGPRs per lane.
__global__ __launch_bounds__(256) void moe_fused_kernel(
    const __bf16* __restrict__ xbf,  const __bf16* __restrict__ W1t,
    const __bf16* __restrict__ W2t,  const float* __restrict__ b1,
    const float* __restrict__ b2,    const int* __restrict__ pair_e,
    const float* __restrict__ pair_w, float* __restrict__ out) {
  extern __shared__ char smem[];
  __bf16* x_lds = (__bf16*)smem;                                // [TILE_S][PX]
  __bf16* h_lds = (__bf16*)(smem + (size_t)TILE_S * PX * 2);    // [TILE_S][PH]

  const int tid = threadIdx.x;
  const int wv  = tid >> 5;          // wave 0..7
  const int lane = tid & 31;
  const int l16 = lane & 15;
  const int lh  = lane >> 4;         // lane half
  const int b   = blockIdx.y;
  const int s0  = blockIdx.x * TILE_S;

  // Stage x tile (bf16) into LDS once; shared by both pairs.
  const __bf16* xsrc = xbf + ((size_t)b * S_ + s0) * D_;
  for (int i = tid; i < TILE_S * (D_ / 8); i += 256) {
    const int r = i >> 7, c = (i & 127) * 8;
    *(bf16x8*)(x_lds + r * PX + c) = *(const bf16x8*)(xsrc + (size_t)r * D_ + c);
  }
  __syncthreads();

  v8f acc[16];                       // acc[dt] = row-block 0, acc[8+dt] = row-block 1
  #pragma unroll
  for (int j = 0; j < 16; ++j) acc[j] = (v8f){0, 0, 0, 0, 0, 0, 0, 0};
  float bias2[8];
  #pragma unroll
  for (int j = 0; j < 8; ++j) bias2[j] = 0.f;

  for (int kp = 0; kp < TOPK_; ++kp) {
    const int   p = b * TOPK_ + kp;
    const int   e = pair_e[p];
    const float w = pair_w[p];
    if (w != 0.f) {                  // block-uniform branch (dropped/zero-gate pair)
      const __bf16* W1e = W1t + (size_t)e * H_ * D_;   // [hh][d]
      const __bf16* W2e = W2t + (size_t)e * D_ * H_;   // [d][hh]
      const float*  b1e = b1 + e * H_;
      for (int ch = 0; ch < H_ / HCHUNK; ++ch) {
        const int h0 = ch * HCHUNK;
        // ---- GEMM1: h^T = W1^T * x^T, +b1, GELU, *gate, -> bf16 LDS ----
        // Each wave: full h-tile; one A-frag feeds both s-tile accumulators.
        for (int ht = wv; ht < HCHUNK / 16; ht += 8) {
          const int hh0 = h0 + ht * 16;
          const __bf16* Ap  = W1e + (size_t)(hh0 + l16) * D_ + lh * 8;    // A: M=hh
          const __bf16* Bp0 = x_lds + l16 * PX + lh * 8;                  // B: N=s (st=0)
          const __bf16* Bp1 = x_lds + (16 + l16) * PX + lh * 8;           // B: N=s (st=1)
          v8f c0 = (v8f){0, 0, 0, 0, 0, 0, 0, 0};
          v8f c1 = (v8f){0, 0, 0, 0, 0, 0, 0, 0};
          #pragma unroll 4
          for (int kb = 0; kb < D_; kb += 32) {
            const bf16x16 a = frag_a(Ap + kb);
            c0 = wmma_bf16(a, frag_b(Bp0 + kb), c0);
            c1 = wmma_bf16(a, frag_b(Bp1 + kb), c1);
          }
          const float* bp = b1e + hh0 + lh * 8;   // lane's 8 consecutive h-cols
          bf16x8 hv0, hv1;
          #pragma unroll
          for (int v = 0; v < 8; ++v) {
            hv0[v] = (__bf16)(gelu_exact(c0[v] + bp[v]) * w);  // fold gate into h
            hv1[v] = (__bf16)(gelu_exact(c1[v] + bp[v]) * w);
          }
          *(bf16x8*)(h_lds + l16 * PH + (ht * 16 + lh * 8)) = hv0;
          *(bf16x8*)(h_lds + (16 + l16) * PH + (ht * 16 + lh * 8)) = hv1;
        }
        __syncthreads();
        // ---- GEMM2: acc += h * W2^T (K = HCHUNK); B-frag reused for both row-blocks ----
        for (int kb = 0; kb < HCHUNK; kb += 32) {
          const bf16x16 a0 = frag_a(h_lds + l16 * PH + lh * 8 + kb);
          const bf16x16 a1 = frag_a(h_lds + (16 + l16) * PH + lh * 8 + kb);
          #pragma unroll
          for (int dt = 0; dt < 8; ++dt) {
            const int d = (wv * 8 + dt) * 16 + l16;                       // B: N=d
            const bf16x16 bb = frag_b(W2e + (size_t)d * H_ + h0 + lh * 8 + kb);
            acc[dt]     = wmma_bf16(a0, bb, acc[dt]);
            acc[8 + dt] = wmma_bf16(a1, bb, acc[8 + dt]);
          }
        }
        __syncthreads();
      }
      #pragma unroll
      for (int dt = 0; dt < 8; ++dt)
        bias2[dt] += w * b2[e * D_ + (wv * 8 + dt) * 16 + l16];
    }
  }
  // ---- store y = acc + sum_p w_p*b2[e_p]  (C layout: M = lh*8+v, N = l16) ----
  #pragma unroll
  for (int srb = 0; srb < 2; ++srb) {
    #pragma unroll
    for (int dt = 0; dt < 8; ++dt) {
      const int d = (wv * 8 + dt) * 16 + l16;
      const float bb = bias2[dt];
      #pragma unroll
      for (int v = 0; v < 8; ++v) {
        const int s = s0 + srb * 16 + lh * 8 + v;
        out[((size_t)b * S_ + s) * D_ + d] = acc[srb * 8 + dt][v] + bb;
      }
    }
  }
}

// ---------------- launch ----------------
extern "C" void kernel_launch(void* const* d_in, const int* in_sizes, int n_in,
                              void* d_out, int out_size, void* d_ws, size_t ws_size,
                              hipStream_t stream) {
  (void)in_sizes; (void)n_in; (void)out_size; (void)ws_size;
  const float* router_input = (const float*)d_in[0];
  const float* x     = (const float*)d_in[1];
  const float* noise = (const float*)d_in[2];
  const float* Wg    = (const float*)d_in[3];
  const float* bg    = (const float*)d_in[4];
  const float* Wn    = (const float*)d_in[5];
  const float* bn    = (const float*)d_in[6];
  const float* W1    = (const float*)d_in[7];
  const float* b1    = (const float*)d_in[8];
  const float* W2    = (const float*)d_in[9];
  const float* b2    = (const float*)d_in[10];
  float* out = (float*)d_out;

  char* ws = (char*)d_ws;
  float*  ri     = (float*)(ws + WS_RI);
  int*    pair_e = (int*)(ws + WS_PAIR_E);
  float*  pair_w = (float*)(ws + WS_PAIR_W);
  __bf16* xbf    = (__bf16*)(ws + WS_XBF);
  __bf16* W1t    = (__bf16*)(ws + WS_W1T);
  __bf16* W2t    = (__bf16*)(ws + WS_W2T);

  moe_mean_kernel<<<B_, 256, 0, stream>>>(router_input, ri);
  moe_router_kernel<<<1, 32, 0, stream>>>(ri, noise, Wg, bg, Wn, bn, pair_e, pair_w);
  moe_cvt_bf16_kernel<<<4096, 256, 0, stream>>>(x, xbf, (size_t)B_ * S_ * D_);
  // W1 [E][D][H] -> W1t [E][H][D]
  moe_transpose_cvt_kernel<<<dim3(H_ / 32, D_ / 32, E_), dim3(32, 8), 0, stream>>>(W1, W1t, D_, H_);
  // W2 [E][H][D] -> W2t [E][D][H]
  moe_transpose_cvt_kernel<<<dim3(D_ / 32, H_ / 32, E_), dim3(32, 8), 0, stream>>>(W2, W2t, H_, D_);

  (void)hipFuncSetAttribute(reinterpret_cast<const void*>(moe_fused_kernel),
                            hipFuncAttributeMaxDynamicSharedMemorySize, SMEM_BYTES);
  moe_fused_kernel<<<dim3(S_ / TILE_S, B_), 256, SMEM_BYTES, stream>>>(
      xbf, W1t, W2t, b1, b2, pair_e, pair_w, out);
}
